// MyModel_79285096284206
// MI455X (gfx1250) — matrix operand order repaired and checked by reading, not compiled, not run on hardware
//
#include <hip/hip_runtime.h>
#include <hip/hip_bf16.h>

// ---------------------------------------------------------------------------
// MI455X (gfx1250) fused 3-layer ReLU-LSTM + dense head.
//
// Roofline reasoning: input projections are ~55 GFLOP of embarrassingly
// parallel GEMM -> bf16 WMMA (v_wmma_f32_16x16x32_bf16). The recurrence is
// 512 *sequential* steps of tiny GEMMs -> persistent kernel, 16 batch rows
// per workgroup, U held in VGPRs as pre-swizzled B-fragments for the whole
// time loop, h/c state in LDS, per-step global traffic limited to the
// precomputed zx row (prefetched) + bf16 h writeback. zx round-trip is only
// ~2.7 GB total (~0.12 ms at 23.3 TB/s HBM), so precomputing it (as the
// reference does) is the right data-movement path.
// ---------------------------------------------------------------------------

typedef __attribute__((ext_vector_type(16))) __bf16 v16bf;
typedef __attribute__((ext_vector_type(8)))  float  v8f;

union Frag { uint4 q[2]; v16bf v; };   // 32 bytes = 16 bf16 = one wmma A/B operand

constexpr int BATCH = 2048;
constexpr int TSEQ  = 512;
constexpr long MROWS = (long)BATCH * TSEQ;   // 1,048,576 rows, multiple of 16

// ---------------------------------------------------------------------------
// Convert f32 weight [Din x G] into bf16 laid out in WMMA B-fragment order:
// linear index = ((kc*G + n)*32 + kk), k = kc*32+kk (zero-padded to KPAD).
// A lane building a B-frag then reads 16 *contiguous* bf16 (32B) per k-chunk.
// ---------------------------------------------------------------------------
__global__ void swizzle_kernel(const float* __restrict__ W, __bf16* __restrict__ out,
                               int Din, int KPAD, int G) {
  int idx = blockIdx.x * blockDim.x + threadIdx.x;
  int total = KPAD * G;
  if (idx >= total) return;
  int kc  = idx / (G * 32);
  int rem = idx - kc * (G * 32);
  int n   = rem >> 5;
  int kk  = rem & 31;
  int k   = kc * 32 + kk;
  float v = (k < Din) ? W[(size_t)k * G + n] : 0.f;
  out[idx] = (__bf16)v;
}

// ---------------------------------------------------------------------------
// zx = in @ W + bias   for all B*T rows.  A staged in LDS (bf16, K padded),
// B frags from pre-swizzled bf16 weights, f32 accumulate via WMMA.
// ---------------------------------------------------------------------------
template<int DIN, int KPAD, int G, bool IN_BF16>
__global__ void __launch_bounds__(128)
proj_kernel(const void* __restrict__ in_, const __bf16* __restrict__ Wswz,
            const float* __restrict__ bias, float* __restrict__ zx) {
  constexpr int WAVES = 4;
  constexpr int GT = G / 16;
  const int tid  = threadIdx.x;
  const int wave = tid >> 5;
  const int lane = tid & 31;
  const int l15  = lane & 15;
  const int hi   = lane >> 4;            // wave32: half-wave selector
  const long row0 = (long)blockIdx.x * 16;

  __shared__ __align__(16) __bf16 a_lds[16][KPAD];

  // Stage + convert the 16xDIN A tile into LDS (zero-padded K).
  for (int i = tid; i < 16 * KPAD; i += 128) {
    int r = i / KPAD, k = i - r * KPAD;
    float v = 0.f;
    if (k < DIN) {
      if (IN_BF16) v = (float)((const __bf16*)in_)[(row0 + r) * DIN + k];
      else         v = ((const float*)in_)[(row0 + r) * DIN + k];
    }
    a_lds[r][k] = (__bf16)v;
  }
  __syncthreads();

  const int nt = blockIdx.y * WAVES + wave;   // wave-uniform -> EXEC all-ones
  if (nt < GT) {
    v8f acc = {};
#pragma unroll
    for (int kc = 0; kc < KPAD / 32; ++kc) {
      Frag a, b;
      // A 16x32 frag, contiguous k order shared with B: lane half selects K base {0,16}; 16 contiguous elems.
      const __bf16* ha = &a_lds[l15][kc * 32 + hi * 16];
      a.q[0] = *(const uint4*)(ha);
      a.q[1] = *(const uint4*)(ha + 8);
      const __bf16* wb = Wswz + ((size_t)(kc * G + nt * 16 + l15) * 32) + hi * 16;
      b.q[0] = ((const uint4*)wb)[0];
      b.q[1] = ((const uint4*)wb)[1];
      acc = __builtin_amdgcn_wmma_f32_16x16x32_bf16(false, a.v, false, b.v,
                                                    (short)0, acc, false, false);
    }
    const int col = nt * 16 + l15;
    const float bc = bias[col];
#pragma unroll
    for (int r = 0; r < 8; ++r) {           // C/D layout: VGPR r -> M = r + 8*hi
      long row = row0 + r + hi * 8;
      zx[(size_t)row * G + col] = acc[r] + bc;
    }
  }
}

// ---------------------------------------------------------------------------
// Persistent recurrence: one block = 16 batch rows for all T steps.
// U lives in VGPRs as B-fragments for the entire time loop; h (bf16) and
// c (f32) live in LDS.  G = 4H, HPAD = K padded to multiple of 32.
// ---------------------------------------------------------------------------
template<int H, int HPAD, int G, int WAVES, int TPW>
__global__ void __launch_bounds__(WAVES * 32)
rec_kernel(const float* __restrict__ zx, const __bf16* __restrict__ Uswz,
           __bf16* __restrict__ h_out) {
  constexpr int NT  = WAVES * 32;
  constexpr int KCH = HPAD / 32;
  static_assert(G / 16 == WAVES * TPW, "tile coverage");

  const int tid  = threadIdx.x;
  const int wave = tid >> 5;
  const int lane = tid & 31;
  const int l15  = lane & 15;
  const int hi   = lane >> 4;
  const long b0  = (long)blockIdx.x * 16;

  __shared__ __align__(16) __bf16 h_lds[16][HPAD];
  __shared__ float z_lds[16][G];
  __shared__ float c_lds[16][H];

  for (int i = tid; i < 16 * HPAD; i += NT) (&h_lds[0][0])[i] = (__bf16)0.f;
  for (int i = tid; i < 16 * H;    i += NT) (&c_lds[0][0])[i] = 0.f;

  // Preload recurrent weights as WMMA B-fragments (resident for all 512 steps).
  Frag bfrag[TPW][KCH];
#pragma unroll
  for (int tp = 0; tp < TPW; ++tp) {
    const int ntile = wave * TPW + tp;
#pragma unroll
    for (int kc = 0; kc < KCH; ++kc) {
      const __bf16* ub = Uswz + ((size_t)(kc * G + ntile * 16 + l15) * 32) + hi * 16;
      bfrag[tp][kc].q[0] = ((const uint4*)ub)[0];
      bfrag[tp][kc].q[1] = ((const uint4*)ub)[1];
    }
  }
  __syncthreads();

  for (int t = 0; t < TSEQ; ++t) {
    // ---- z = h_prev @ U  (WMMA, A-frags from LDS via ds_load_b128) ----
#pragma unroll
    for (int tp = 0; tp < TPW; ++tp) {
      const int ntile = wave * TPW + tp;
      v8f acc = {};
#pragma unroll
      for (int kc = 0; kc < KCH; ++kc) {
        Frag a;
        const __bf16* ha = &h_lds[l15][kc * 32 + hi * 16];
        a.q[0] = *(const uint4*)(ha);
        a.q[1] = *(const uint4*)(ha + 8);
        acc = __builtin_amdgcn_wmma_f32_16x16x32_bf16(false, a.v, false,
                                                      bfrag[tp][kc].v,
                                                      (short)0, acc, false, false);
      }
      const int col = ntile * 16 + l15;
#pragma unroll
      for (int r = 0; r < 8; ++r) z_lds[r + hi * 8][col] = acc[r];
    }
    __syncthreads();

    // ---- gates (divergence-free: 16*H is a multiple of NT's stride loop) ----
    for (int i = tid; i < 16 * H; i += NT) {
      const int m = i / H, j = i - m * H;
      const float* zr = zx + (((size_t)(b0 + m) * TSEQ) + t) * G;
      float zi = z_lds[m][j]         + zr[j];
      float zf = z_lds[m][H + j]     + zr[H + j];
      float zg = z_lds[m][2 * H + j] + zr[2 * H + j];
      float zo = z_lds[m][3 * H + j] + zr[3 * H + j];
      float gi = 1.f / (1.f + __expf(-zi));
      float gf = 1.f / (1.f + __expf(-zf));
      float gg = fmaxf(zg, 0.f);
      float go = 1.f / (1.f + __expf(-zo));
      float c  = gf * c_lds[m][j] + gi * gg;
      c_lds[m][j] = c;
      float h = go * fmaxf(c, 0.f);
      h_lds[m][j] = (__bf16)h;
      h_out[(((size_t)(b0 + m) * TSEQ) + t) * H + j] = (__bf16)h;
      if (t + 1 < TSEQ) __builtin_prefetch(zr + G, 0, 0);  // global_prefetch_b8
    }
    __syncthreads();
  }
}

// ---------------------------------------------------------------------------
// out = h3 @ Wd + bd    (H=40, trivial dot per row; bandwidth-bound)
// ---------------------------------------------------------------------------
__global__ void dense_kernel(const __bf16* __restrict__ h3, const float* __restrict__ Wd,
                             const float* __restrict__ bd, float* __restrict__ out) {
  long idx = (long)blockIdx.x * blockDim.x + threadIdx.x;
  if (idx >= MROWS) return;
  const __bf16* hr = h3 + idx * 40;
  float s = bd[0];
#pragma unroll
  for (int j = 0; j < 40; ++j) s += (float)hr[j] * Wd[j];
  out[idx] = s;
}

// ---------------------------------------------------------------------------
extern "C" void kernel_launch(void* const* d_in, const int* in_sizes, int n_in,
                              void* d_out, int out_size, void* d_ws, size_t ws_size,
                              hipStream_t stream) {
  (void)in_sizes; (void)n_in; (void)out_size; (void)ws_size;
  const float* x  = (const float*)d_in[0];
  const float* W1 = (const float*)d_in[1];
  const float* U1 = (const float*)d_in[2];
  const float* b1 = (const float*)d_in[3];
  const float* W2 = (const float*)d_in[4];
  const float* U2 = (const float*)d_in[5];
  const float* b2 = (const float*)d_in[6];
  const float* W3 = (const float*)d_in[7];
  const float* U3 = (const float*)d_in[8];
  const float* b3 = (const float*)d_in[9];
  const float* Wd = (const float*)d_in[10];
  const float* bd = (const float*)d_in[11];
  float* out = (float*)d_out;

  char* base = (char*)d_ws;
  size_t off = 0;
  auto take = [&](size_t elems, size_t elsz) -> char* {
    char* p = base + off;
    off += (elems * elsz + 255) & ~(size_t)255;
    return p;
  };
  __bf16* W1s = (__bf16*)take(32 * 160, 2);
  __bf16* U1s = (__bf16*)take(64 * 160, 2);
  __bf16* W2s = (__bf16*)take(64 * 320, 2);
  __bf16* U2s = (__bf16*)take(96 * 320, 2);
  __bf16* W3s = (__bf16*)take(96 * 160, 2);
  __bf16* U3s = (__bf16*)take(64 * 160, 2);
  __bf16* h1  = (__bf16*)take((size_t)MROWS * 40, 2);
  __bf16* h2  = (__bf16*)take((size_t)MROWS * 80, 2);
  __bf16* h3  = (__bf16*)take((size_t)MROWS * 40, 2);
  float*  zx  = (float*) take((size_t)MROWS * 320, 4);   // reused by each layer

  auto swz = [&](const float* W, __bf16* o, int Din, int KPAD, int G) {
    int total = KPAD * G;
    swizzle_kernel<<<(total + 255) / 256, 256, 0, stream>>>(W, o, Din, KPAD, G);
  };
  swz(W1, W1s,  5, 32, 160);
  swz(U1, U1s, 40, 64, 160);
  swz(W2, W2s, 40, 64, 320);
  swz(U2, U2s, 80, 96, 320);
  swz(W3, W3s, 80, 96, 160);
  swz(U3, U3s, 40, 64, 160);

  const int MT = (int)(MROWS / 16);   // 65536 M tiles

  // Layer 1: H=40, G=160 (10 N-tiles)
  proj_kernel<5, 32, 160, false><<<dim3(MT, 3), 128, 0, stream>>>(x, W1s, b1, zx);
  rec_kernel<40, 64, 160, 10, 1><<<dim3(BATCH / 16), 320, 0, stream>>>(zx, U1s, h1);

  // Layer 2: H=80, G=320 (20 N-tiles)
  proj_kernel<40, 64, 320, true><<<dim3(MT, 5), 128, 0, stream>>>(h1, W2s, b2, zx);
  rec_kernel<80, 96, 320, 10, 2><<<dim3(BATCH / 16), 320, 0, stream>>>(zx, U2s, h2);

  // Layer 3: H=40, G=160
  proj_kernel<80, 96, 160, true><<<dim3(MT, 3), 128, 0, stream>>>(h2, W3s, b3, zx);
  rec_kernel<40, 64, 160, 10, 1><<<dim3(BATCH / 16), 320, 0, stream>>>(zx, U3s, h3);

  // Dense head
  dense_kernel<<<(int)((MROWS + 255) / 256), 256, 0, stream>>>(h3, Wd, bd, out);
}
